// NerfRender_10368051052709
// MI455X (gfx1250) — matrix operand order repaired and checked by reading, not compiled, not run on hardware
//
#include <hip/hip_runtime.h>

typedef __attribute__((ext_vector_type(16))) _Float16 v16h;
typedef __attribute__((ext_vector_type(8)))  _Float16 v8h;
typedef __attribute__((ext_vector_type(8)))  float    v8f;

#define TILE_M   128      // 2 rays x 64 samples per workgroup
#define ASTRIDE  328      // LDS activation row stride in halves (bank-friendly, >=320)

// ---- fragment table (units: fragments of 512 halves = 1024 B) ----
enum : int {
  OFF_FWD0 = 0,    OFF_FWD1 = 32,   OFF_FWD2 = 160,  OFF_FWD3 = 288,
  OFF_FWD4 = 416,  OFF_FWD5 = 576,  OFF_FWD6 = 704,  OFF_FWD7 = 832,
  OFF_CW0  = 960,
  OFF_BWD0 = 1040, OFF_BWD1 = 1072, OFF_BWD2 = 1200, OFF_BWD3 = 1328,
  OFF_BWD4 = 1456, OFF_BWD5 = 1616, OFF_BWD6 = 1744, OFF_BWD7 = 1872,
  FRAG_TOTAL = 2000
};

// ---- flat output offsets (floats): C(2048*3), Dp(2048), wi(2048*64), t(2048*64)
enum : int { OUT_C = 0, OUT_DP = 6144, OUT_WI = 8192, OUT_T = 139264 };

struct NerfParams {
  const float* ray;
  const float* t;
  const float* db[8];
  const float* Wy;  const float* by;
  const float* cW0; const float* cb0;
  const float* cW1; const float* cb1;
  const _Float16* frags;
  float* out;
};

// =====================================================================
// Prep: repack an (R,C) row-major f32 weight into f16 WMMA B-fragments.
// Fragment (kt,nt): lane l holds element j -> B[kt*32 + (l>=16?16:0)+j][nt*16 + (l&15)]
// transpose=1 stores B = W^T (B[k][n] = W[n][k]).
// =====================================================================
__global__ void prep_frags(const float* __restrict__ W, _Float16* __restrict__ dst,
                           int R, int C, int Ktiles, int Ntiles, int transpose)
{
  const int frag = blockIdx.x;             // kt*Ntiles + nt
  const int kt = frag / Ntiles, nt = frag % Ntiles;
  _Float16* d = dst + (size_t)frag * 512;
  for (int e = threadIdx.x; e < 512; e += blockDim.x) {
    const int lane = e >> 4, j = e & 15;
    const int k = kt * 32 + ((lane >> 4) << 4) + j;
    const int n = nt * 16 + (lane & 15);
    float v = 0.0f;
    if (!transpose) { if (k < R && n < C) v = W[(size_t)k * C + n]; }
    else            { if (n < R && k < C) v = W[(size_t)n * C + k]; }
    d[e] = (_Float16)v;
  }
}

// =====================================================================
// One tile-GEMM layer: 128xK (LDS f16) @ KxN (global f16 frags) -> 128xN
// Wave w owns rows [16w,16w+16). Two independent accumulator chains keep
// the XDL pipe fed; B fragments are preloaded as one clause and the next
// N-strip is prefetched into WGP$ while the current strip computes.
// Store paths: raw f32, relu+mask-record, mask-apply (backward), plain f16.
// =====================================================================
template<int KTILES>
__device__ void tile_gemm(const _Float16* __restrict__ wfrag,
                          const _Float16* __restrict__ actIn,
                          _Float16* actOut,
                          float* outF32, int strideF32,
                          const float* __restrict__ bias,
                          int Ntiles,
                          unsigned int* maskRec,
                          const unsigned int* maskApp, int maskNT)
{
  const int lane  = threadIdx.x & 31;
  const int wave  = threadIdx.x >> 5;   // 0..7 -> M tile
  const int g     = lane >> 4;
  const int col   = lane & 15;
  const int mbase = wave * 16 + g * 8;

  // A fragments: lane holds row (lane&15) of this wave's band; element j -> k = kt*32 + 16*(lane>>4) + j
  v16h a[KTILES];
  {
    const _Float16* arow = actIn + (size_t)(wave * 16 + col) * ASTRIDE;
#pragma unroll
    for (int kt = 0; kt < KTILES; ++kt) {
      const int kb = kt * 32 + g * 16;
      v8h lo = *(const v8h*)(arow + kb);
      v8h hi = *(const v8h*)(arow + kb + 8);
      v16h av;
#pragma unroll
      for (int j = 0; j < 8; ++j) { av[j] = lo[j]; av[j + 8] = hi[j]; }
      a[kt] = av;
    }
  }

  for (int nt = 0; nt < Ntiles; ++nt) {
    const int n = nt * 16 + col;

    // B fragments for this strip: one clause of b128 loads
    const _Float16* wp = wfrag + (size_t)nt * 512 + (size_t)lane * 16;
    v16h b[KTILES];
#pragma unroll
    for (int kt = 0; kt < KTILES; ++kt)
      b[kt] = *(const v16h*)(wp + (size_t)kt * Ntiles * 512);

    // warm WGP$ for the next strip while this one computes
    if (nt + 1 < Ntiles) {
      const _Float16* wpn = wp + 512;
#pragma unroll
      for (int kt = 0; kt < KTILES; ++kt)
        __builtin_prefetch((const void*)(wpn + (size_t)kt * Ntiles * 512), 0, 1);
    }

    const float bv = bias ? bias[n] : 0.0f;
    v8f acc0, acc1;
#pragma unroll
    for (int r = 0; r < 8; ++r) { acc0[r] = bv; acc1[r] = 0.0f; }

    // two independent accumulation chains (even/odd kt) for XDL ILP
#pragma unroll
    for (int kt = 0; kt < KTILES; kt += 2) {
      acc0 = __builtin_amdgcn_wmma_f32_16x16x32_f16(false, a[kt], false, b[kt],
                                                    (short)0, acc0, false, false);
      if (kt + 1 < KTILES)
        acc1 = __builtin_amdgcn_wmma_f32_16x16x32_f16(false, a[kt + 1], false, b[kt + 1],
                                                      (short)0, acc1, false, false);
    }
    v8f acc;
#pragma unroll
    for (int r = 0; r < 8; ++r) acc[r] = acc0[r] + acc1[r];

    if (outF32) {
#pragma unroll
      for (int r = 0; r < 8; ++r)
        outF32[(size_t)(mbase + r) * strideF32 + n] = acc[r];
    } else if (maskRec) {                       // forward: relu + record mask bits
      unsigned int bits = 0;
#pragma unroll
      for (int r = 0; r < 8; ++r) {
        float v = acc[r];
        if (v > 0.0f) bits |= (1u << r); else v = 0.0f;
        actOut[(size_t)(mbase + r) * ASTRIDE + n] = (_Float16)v;
      }
      atomicOr(&maskRec[n * 4 + (mbase >> 5)], bits << (mbase & 31));
    } else if (maskApp && nt < maskNT) {        // backward: gate by stored relu mask
      const unsigned int wb = maskApp[n * 4 + (mbase >> 5)] >> (mbase & 31);
#pragma unroll
      for (int r = 0; r < 8; ++r) {
        float v = ((wb >> r) & 1u) ? acc[r] : 0.0f;
        actOut[(size_t)(mbase + r) * ASTRIDE + n] = (_Float16)v;
      }
    } else {
#pragma unroll
      for (int r = 0; r < 8; ++r)
        actOut[(size_t)(mbase + r) * ASTRIDE + n] = (_Float16)acc[r];
    }
  }
}

// =====================================================================
// Main kernel: one workgroup (256 thr / 8 waves) = 2 rays = 128 points.
// =====================================================================
__global__ void __launch_bounds__(256)
nerf_tile_kernel(NerfParams P)
{
  extern __shared__ char smem[];
  _Float16*     bufA  = (_Float16*)smem;                          // 128 x 328 halves
  _Float16*     bufB  = bufA + (size_t)TILE_M * ASTRIDE;          // 128 x 328 halves
  float*        c1buf = (float*)(bufB + (size_t)TILE_M * ASTRIDE);// 128 x 132 f32
  unsigned int* maskW = (unsigned int*)(c1buf + TILE_M * 132);    // 8 x 256 x 4 words
  float*        tbuf  = (float*)(maskW + 8 * 256 * 4);            // 128
  float*        sigb  = tbuf + TILE_M;                            // 128
  float*        xbuf  = sigb + TILE_M;                            // 128*3
  float*        nbuf  = xbuf + TILE_M * 3;                        // 128*3
  float*        colb  = nbuf + TILE_M * 3;                        // 128*3

  const int tid  = threadIdx.x;
  const int ray0 = blockIdx.x * 2;
  const _Float16* F = P.frags;

  for (int i = tid; i < 8 * 256 * 4; i += 256) maskW[i] = 0;
  if (tid < 128) tbuf[tid] = P.t[(size_t)ray0 * 64 + tid];
  __syncthreads();

  // ---- bitonic sort each ray's 64 t-samples ascending ----
  for (int k = 2; k <= 64; k <<= 1) {
    for (int j = k >> 1; j > 0; j >>= 1) {
      if (tid < 128) {
        const int i = tid & 63, base = tid & 64;
        const int p = i ^ j;
        if (p > i) {
          float va = tbuf[base + i], vb = tbuf[base + p];
          const bool up = ((i & k) == 0);
          if (up ? (va > vb) : (va < vb)) { tbuf[base + i] = vb; tbuf[base + p] = va; }
        }
      }
      __syncthreads();
    }
  }
  if (tid < 128) P.out[OUT_T + (size_t)ray0 * 64 + tid] = tbuf[tid];

  // ---- x = o + t*d ; e = pe(x,10) -> bufA[0:64) and e replica at cols [256,320) ----
  if (tid < 128) {
    const int r = tid >> 6;
    const float* rp = P.ray + (size_t)(ray0 + r) * 6;
    const float ts = tbuf[tid];
    float xc[3];
#pragma unroll
    for (int c = 0; c < 3; ++c) { xc[c] = rp[c] + ts * rp[3 + c]; xbuf[tid * 3 + c] = xc[c]; }
    float e[64];
    e[0] = xc[0]; e[1] = xc[1]; e[2] = xc[2]; e[63] = 0.0f;
#pragma unroll
    for (int c = 0; c < 3; ++c) {
      float f = 1.0f;
#pragma unroll
      for (int l = 0; l < 10; ++l) {
        const float ang = xc[c] * f;
        e[3 + c * 10 + l]  = __sinf(ang);
        e[33 + c * 10 + l] = __cosf(ang);
        f *= 2.0f;
      }
    }
    _Float16* rowA = bufA + (size_t)tid * ASTRIDE;
    _Float16* rowB = bufB + (size_t)tid * ASTRIDE;
#pragma unroll
    for (int i2 = 0; i2 < 64; ++i2) {
      const _Float16 h = (_Float16)e[i2];
      rowA[i2] = h; rowA[256 + i2] = h; rowB[256 + i2] = h;
    }
  }
  __syncthreads();

  // ---- forward density MLP (skip concat at layer 4 via e at cols 256..319) ----
  tile_gemm<2 >(F + (size_t)OFF_FWD0 * 512, bufA, bufB, nullptr, 0, P.db[0], 16, maskW + 0 * 1024, nullptr, 0); __syncthreads();
  tile_gemm<8 >(F + (size_t)OFF_FWD1 * 512, bufB, bufA, nullptr, 0, P.db[1], 16, maskW + 1 * 1024, nullptr, 0); __syncthreads();
  tile_gemm<8 >(F + (size_t)OFF_FWD2 * 512, bufA, bufB, nullptr, 0, P.db[2], 16, maskW + 2 * 1024, nullptr, 0); __syncthreads();
  tile_gemm<8 >(F + (size_t)OFF_FWD3 * 512, bufB, bufA, nullptr, 0, P.db[3], 16, maskW + 3 * 1024, nullptr, 0); __syncthreads();
  tile_gemm<10>(F + (size_t)OFF_FWD4 * 512, bufA, bufB, nullptr, 0, P.db[4], 16, maskW + 4 * 1024, nullptr, 0); __syncthreads();
  tile_gemm<8 >(F + (size_t)OFF_FWD5 * 512, bufB, bufA, nullptr, 0, P.db[5], 16, maskW + 5 * 1024, nullptr, 0); __syncthreads();
  tile_gemm<8 >(F + (size_t)OFF_FWD6 * 512, bufA, bufB, nullptr, 0, P.db[6], 16, maskW + 6 * 1024, nullptr, 0); __syncthreads();
  tile_gemm<8 >(F + (size_t)OFF_FWD7 * 512, bufB, bufA, nullptr, 0, P.db[7], 16, maskW + 7 * 1024, nullptr, 0); __syncthreads();
  // h7 now in bufA cols [0,256)

  // ---- y = h7 . Wy + by ; sigma = sigmoid(-10 y) ----
  if (tid < 128) {
    const _Float16* h = bufA + (size_t)tid * ASTRIDE;
    float acc = P.by[0];
    for (int nn = 0; nn < 256; ++nn) acc += (float)h[nn] * P.Wy[nn];
    sigb[tid] = 1.0f / (1.0f + __expf(10.0f * acc));
  }
  __syncthreads();

  // ---- assemble color-net input [x, 0(n), h, ped] into bufB (n-rows added later) ----
  for (int idx = tid; idx < TILE_M * 320; idx += 256) {
    const int p = idx / 320, c = idx % 320;
    const int r = p >> 6;
    float v = 0.0f;
    if (c < 3)        v = xbuf[p * 3 + c];
    else if (c < 6)   v = 0.0f;                                   // normal (linear; added post-backward)
    else if (c < 262) v = (float)bufA[(size_t)p * ASTRIDE + (c - 6)];
    else if (c < 265) v = P.ray[(size_t)(ray0 + r) * 6 + 3 + (c - 262)];
    else if (c < 277) { const int q = c - 265; v = __sinf(P.ray[(size_t)(ray0 + r) * 6 + 3 + (q >> 2)] * (float)(1 << (q & 3))); }
    else if (c < 289) { const int q = c - 277; v = __cosf(P.ray[(size_t)(ray0 + r) * 6 + 3 + (q >> 2)] * (float)(1 << (q & 3))); }
    bufB[(size_t)p * ASTRIDE + c] = (_Float16)v;
  }
  __syncthreads();

  // ---- c1pre = cin @ cW0 + cb0 (f32 in LDS, relu deferred) ----
  tile_gemm<10>(F + (size_t)OFF_CW0 * 512, bufB, nullptr, c1buf, 132, P.cb0, 8, nullptr, nullptr, 0);
  __syncthreads();

  // ---- backward: g7 = Wy * mask7 ----
  for (int idx = tid; idx < TILE_M * 256; idx += 256) {
    const int p = idx >> 8, nn = idx & 255;
    const unsigned int mb = (maskW[7 * 1024 + nn * 4 + (p >> 5)] >> (p & 31)) & 1u;
    bufA[(size_t)p * ASTRIDE + nn] = (_Float16)(mb ? P.Wy[nn] : 0.0f);
  }
  __syncthreads();

  // ---- backward chain g_{i-1} = (g_i . m_i) @ W_i^T ----
  tile_gemm<8>(F + (size_t)OFF_BWD7 * 512, bufA, bufB, nullptr, 0, nullptr, 16, nullptr, maskW + 6 * 1024, 16); __syncthreads(); // g6
  tile_gemm<8>(F + (size_t)OFF_BWD6 * 512, bufB, bufA, nullptr, 0, nullptr, 16, nullptr, maskW + 5 * 1024, 16); __syncthreads(); // g5
  tile_gemm<8>(F + (size_t)OFF_BWD5 * 512, bufA, bufB, nullptr, 0, nullptr, 16, nullptr, maskW + 4 * 1024, 16); __syncthreads(); // g4
  tile_gemm<8>(F + (size_t)OFF_BWD4 * 512, bufB, bufA, nullptr, 0, nullptr, 20, nullptr, maskW + 3 * 1024, 16); __syncthreads(); // [g3 | ge_a]
  tile_gemm<8>(F + (size_t)OFF_BWD3 * 512, bufA, bufB, nullptr, 0, nullptr, 16, nullptr, maskW + 2 * 1024, 16); __syncthreads(); // g2
  tile_gemm<8>(F + (size_t)OFF_BWD2 * 512, bufB, bufA, nullptr, 0, nullptr, 16, nullptr, maskW + 1 * 1024, 16); __syncthreads(); // g1 (ge_a safe at cols 256+)
  tile_gemm<8>(F + (size_t)OFF_BWD1 * 512, bufA, bufB, nullptr, 0, nullptr, 16, nullptr, maskW + 0 * 1024, 16); __syncthreads(); // g0
  tile_gemm<8>(F + (size_t)OFF_BWD0 * 512, bufB, bufB + 256, nullptr, 0, nullptr, 4, nullptr, nullptr, 0);      __syncthreads(); // ge_b

  // ---- PE chain rule -> dx ; n = normalize(dx) ----
  if (tid < 128) {
    float ge[63];
#pragma unroll
    for (int i2 = 0; i2 < 63; ++i2)
      ge[i2] = (float)bufA[(size_t)tid * ASTRIDE + 256 + i2] + (float)bufB[(size_t)tid * ASTRIDE + 256 + i2];
    float dx[3];
#pragma unroll
    for (int c = 0; c < 3; ++c) {
      float s = ge[c], f = 1.0f;
      const float xc = xbuf[tid * 3 + c];
#pragma unroll
      for (int l = 0; l < 10; ++l) {
        const float ang = xc * f;
        s += f * (__cosf(ang) * ge[3 + c * 10 + l] - __sinf(ang) * ge[33 + c * 10 + l]);
        f *= 2.0f;
      }
      dx[c] = s;
    }
    const float nrm = sqrtf(dx[0] * dx[0] + dx[1] * dx[1] + dx[2] * dx[2]) + 1e-6f;
    nbuf[tid * 3 + 0] = dx[0] / nrm; nbuf[tid * 3 + 1] = dx[1] / nrm; nbuf[tid * 3 + 2] = dx[2] / nrm;
  }
  __syncthreads();

  // ---- c1 = relu(c1pre + n-part) ----
  for (int idx = tid; idx < TILE_M * 128; idx += 256) {
    const int p = idx >> 7, nn = idx & 127;
    float v = c1buf[p * 132 + nn]
            + nbuf[p * 3 + 0] * P.cW0[(size_t)3 * 128 + nn]
            + nbuf[p * 3 + 1] * P.cW0[(size_t)4 * 128 + nn]
            + nbuf[p * 3 + 2] * P.cW0[(size_t)5 * 128 + nn];
    c1buf[p * 132 + nn] = v > 0.0f ? v : 0.0f;
  }
  __syncthreads();

  // ---- color = sigmoid(c1 @ cW1 + cb1) ----
  if (tid < 128) {
    float a0 = P.cb1[0], a1 = P.cb1[1], a2 = P.cb1[2];
    for (int nn = 0; nn < 128; ++nn) {
      const float h = c1buf[tid * 132 + nn];
      a0 += h * P.cW1[nn * 3 + 0]; a1 += h * P.cW1[nn * 3 + 1]; a2 += h * P.cW1[nn * 3 + 2];
    }
    colb[tid * 3 + 0] = 1.0f / (1.0f + __expf(-a0));
    colb[tid * 3 + 1] = 1.0f / (1.0f + __expf(-a1));
    colb[tid * 3 + 2] = 1.0f / (1.0f + __expf(-a2));
  }
  __syncthreads();

  // ---- compositing (sequential per ray) ----
  if (tid < 2) {
    float T = 1.0f, C0 = 0.0f, C1 = 0.0f, C2 = 0.0f, Dp = 0.0f;
    for (int s = 0; s < 64; ++s) {
      const int p = tid * 64 + s;
      const float sg = sigb[p];
      const float w = sg * T; T *= (1.0f - sg);
      C0 += w * colb[p * 3 + 0]; C1 += w * colb[p * 3 + 1]; C2 += w * colb[p * 3 + 2];
      Dp += w * tbuf[p];
      P.out[OUT_WI + (size_t)(ray0 + tid) * 64 + s] = w;
    }
    P.out[OUT_C + (size_t)(ray0 + tid) * 3 + 0] = C0;
    P.out[OUT_C + (size_t)(ray0 + tid) * 3 + 1] = C1;
    P.out[OUT_C + (size_t)(ray0 + tid) * 3 + 2] = C2;
    P.out[OUT_DP + ray0 + tid] = Dp;
  }
}

// =====================================================================
extern "C" void kernel_launch(void* const* d_in, const int* in_sizes, int n_in,
                              void* d_out, int out_size, void* d_ws, size_t ws_size,
                              hipStream_t stream)
{
  (void)in_sizes; (void)n_in; (void)out_size; (void)ws_size;
  _Float16* frags = (_Float16*)d_ws;

  struct PD { int w, off, R, C, KT, NT, tr; };
  static const PD pd[17] = {
    // forward B = W            (K = fin padded, N = fout)
    { 2, OFF_FWD0,  63, 256,  2, 16, 0},
    { 4, OFF_FWD1, 256, 256,  8, 16, 0},
    { 6, OFF_FWD2, 256, 256,  8, 16, 0},
    { 8, OFF_FWD3, 256, 256,  8, 16, 0},
    {10, OFF_FWD4, 319, 256, 10, 16, 0},
    {12, OFF_FWD5, 256, 256,  8, 16, 0},
    {14, OFF_FWD6, 256, 256,  8, 16, 0},
    {16, OFF_FWD7, 256, 256,  8, 16, 0},
    {20, OFF_CW0,  289, 128, 10,  8, 0},
    // backward B = W^T         (K = fout, N = fin padded)
    { 2, OFF_BWD0,  63, 256,  8,  4, 1},
    { 4, OFF_BWD1, 256, 256,  8, 16, 1},
    { 6, OFF_BWD2, 256, 256,  8, 16, 1},
    { 8, OFF_BWD3, 256, 256,  8, 16, 1},
    {10, OFF_BWD4, 319, 256,  8, 20, 1},
    {12, OFF_BWD5, 256, 256,  8, 16, 1},
    {14, OFF_BWD6, 256, 256,  8, 16, 1},
    {16, OFF_BWD7, 256, 256,  8, 16, 1},
  };
  for (int i = 0; i < 17; ++i) {
    prep_frags<<<dim3(pd[i].KT * pd[i].NT), dim3(256), 0, stream>>>(
        (const float*)d_in[pd[i].w], frags + (size_t)pd[i].off * 512,
        pd[i].R, pd[i].C, pd[i].KT, pd[i].NT, pd[i].tr);
  }

  NerfParams P;
  P.ray = (const float*)d_in[0];
  P.t   = (const float*)d_in[1];
  for (int i = 0; i < 8; ++i) P.db[i] = (const float*)d_in[3 + 2 * i];
  P.Wy  = (const float*)d_in[18]; P.by  = (const float*)d_in[19];
  P.cW0 = (const float*)d_in[20]; P.cb0 = (const float*)d_in[21];
  P.cW1 = (const float*)d_in[22]; P.cb1 = (const float*)d_in[23];
  P.frags = frags;
  P.out   = (float*)d_out;

  const size_t lds =
      2 * (size_t)TILE_M * ASTRIDE * sizeof(_Float16)   // bufA + bufB
    + (size_t)TILE_M * 132 * sizeof(float)              // c1buf
    + 8 * 256 * 4 * sizeof(unsigned int)                // relu masks
    + (size_t)(TILE_M + TILE_M + TILE_M * 3 + TILE_M * 3 + TILE_M * 3) * sizeof(float);

  nerf_tile_kernel<<<dim3(1024), dim3(256), lds, stream>>>(P);
}
